// DoubleBinaryLinear_44126493999578
// MI455X (gfx1250) — compile-verified, matrix operand order checked by reading, not run on hardware
//
#include <hip/hip_runtime.h>
#include <hip/hip_bf16.h>

// ---------------------------------------------------------------------------
// DoubleBinaryLinear on MI455X (gfx1250):
//   out = ((x*s0) @ B^T * s2) @ A^T * s4 + bias
// Binary +/-1 weights are exact in bf16 -> v_wmma_f32_16x16x32_bf16 core.
// Data movement: global_load_async_to_lds_b128 (ASYNCcnt) + double-buffered
// LDS so the next K-tile streams in behind the WMMA pipeline.
// ---------------------------------------------------------------------------

typedef __attribute__((ext_vector_type(16))) __bf16 v16bf;
typedef __attribute__((ext_vector_type(8)))  float  v8f;

#define K_DIM   4096
#define N_DIM   4096
#define TOKENS  8192   // BATCH*SEQ = 4*2048

constexpr int BM  = 128;
constexpr int BN  = 128;
constexpr int BK  = 64;
constexpr int LDK = BK + 8;   // 144B row pitch: keeps 16B alignment, staggers banks

union Frag16 {
    v16bf v;
    struct { int4 lo; int4 hi; } q;
};

// --- prep: xs = bf16(x * scaling0) -----------------------------------------
__global__ __launch_bounds__(256) void scale_cvt_x(const float* __restrict__ x,
                                                   const float* __restrict__ s,
                                                   __bf16* __restrict__ out) {
    size_t i = ((size_t)blockIdx.x * blockDim.x + threadIdx.x) * 4;
    float4 v  = *(const float4*)(x + i);
    float4 sc = *(const float4*)(s + (i & (size_t)(K_DIM - 1)));
    union { __bf16 h[4]; ushort4 u; } o;
    o.h[0] = (__bf16)(v.x * sc.x);
    o.h[1] = (__bf16)(v.y * sc.y);
    o.h[2] = (__bf16)(v.z * sc.z);
    o.h[3] = (__bf16)(v.w * sc.w);
    *(ushort4*)(out + i) = o.u;
}

// --- prep: bf16(weights) ----------------------------------------------------
__global__ __launch_bounds__(256) void cvt_w(const float* __restrict__ w,
                                             __bf16* __restrict__ out) {
    size_t i = ((size_t)blockIdx.x * blockDim.x + threadIdx.x) * 4;
    float4 v = *(const float4*)(w + i);
    union { __bf16 h[4]; ushort4 u; } o;
    o.h[0] = (__bf16)v.x;
    o.h[1] = (__bf16)v.y;
    o.h[2] = (__bf16)v.z;
    o.h[3] = (__bf16)v.w;
    *(ushort4*)(out + i) = o.u;
}

// Issue one 128x64 bf16 tile pair (A rows, W rows) as async global->LDS copies.
// 256 threads x 4 chunks x 16B = 16KB per matrix; 8 async b128 ops per wave.
__device__ __forceinline__ void issue_tile_async(const __bf16* __restrict__ Amat,
                                                 const __bf16* __restrict__ Wmat,
                                                 int blockM0, int blockN0, int k0,
                                                 unsigned asBase, unsigned bsBase,
                                                 int tid) {
#pragma unroll
    for (int t = 0; t < 4; ++t) {
        const int chunk = tid + t * 256;
        const int row   = chunk >> 3;
        const int col   = (chunk & 7) * 8;
        const unsigned loff = (unsigned)((row * LDK + col) * 2);
        const __bf16* ga = Amat + (size_t)(blockM0 + row) * K_DIM + k0 + col;
        asm volatile("global_load_async_to_lds_b128 %0, %1, off"
                     :: "v"(asBase + loff), "v"(ga) : "memory");
        const __bf16* gw = Wmat + (size_t)(blockN0 + row) * K_DIM + k0 + col;
        asm volatile("global_load_async_to_lds_b128 %0, %1, off"
                     :: "v"(bsBase + loff), "v"(gw) : "memory");
    }
}

// --- main GEMM: D[M=8192, N=4096] = Amat[M,K] * Wmat[N,K]^T -----------------
// EPI==0: store bf16( acc * scale[n] )           (mid activation)
// EPI==1: store f32 ( acc * scale[n] + bias[n] ) (final output)
template <int EPI>
__global__ __launch_bounds__(256, 2)
void gemm_bin_bf16(const __bf16* __restrict__ Amat,
                   const __bf16* __restrict__ Wmat,
                   void* __restrict__ outp,
                   const float* __restrict__ scale,
                   const float* __restrict__ bias) {
    __shared__ __bf16 As[2][BM * LDK];
    __shared__ __bf16 Bs[2][BN * LDK];

    const int tid   = threadIdx.x;
    const int lane  = tid & 31;
    const int wave  = tid >> 5;
    const int waveM = wave & 1;   // 2 waves along M
    const int waveN = wave >> 1;  // 4 waves along N
    const int half  = lane >> 4;  // 0: lanes 0-15, 1: lanes 16-31
    const int r16   = lane & 15;

    const int blockN0 = blockIdx.x * BN;
    const int blockM0 = blockIdx.y * BM;

    // Raw LDS byte offsets (low 32 bits of flat shared pointer).
    const unsigned as0 = (unsigned)(uintptr_t)&As[0][0];
    const unsigned as1 = (unsigned)(uintptr_t)&As[1][0];
    const unsigned bs0 = (unsigned)(uintptr_t)&Bs[0][0];
    const unsigned bs1 = (unsigned)(uintptr_t)&Bs[1][0];

    v8f acc[4][2];
#pragma unroll
    for (int tm = 0; tm < 4; ++tm)
#pragma unroll
        for (int tn = 0; tn < 2; ++tn)
#pragma unroll
            for (int e = 0; e < 8; ++e) acc[tm][tn][e] = 0.0f;

    constexpr int NK = K_DIM / BK;   // 64 K-tiles

    // Prologue: start tile 0 into buffer 0.
    issue_tile_async(Amat, Wmat, blockM0, blockN0, 0, as0, bs0, tid);

    for (int i = 0; i < NK; ++i) {
        const int buf = i & 1;
        // Kick off the next tile into the other buffer while we compute.
        if (i + 1 < NK) {
            issue_tile_async(Amat, Wmat, blockM0, blockN0, (i + 1) * BK,
                             buf ? as0 : as1, buf ? bs0 : bs1, tid);
            // 8 newer async ops in flight; <=8 means tile i has landed (in-order).
            asm volatile("s_wait_asynccnt 0x8" ::: "memory");
        } else {
            asm volatile("s_wait_asynccnt 0x0" ::: "memory");
        }
        __syncthreads();   // tile i resident in LDS for all waves

        const __bf16* __restrict__ Asb = &As[buf][0];
        const __bf16* __restrict__ Bsb = &Bs[buf][0];
#pragma unroll
        for (int kk = 0; kk < BK; kk += 32) {
            const int koff = kk + half * 8; // lanes 0-15: K 0-7/16-23; 16-31: K 8-15/24-31
            Frag16 aF[4], bF[2];
#pragma unroll
            for (int tn = 0; tn < 2; ++tn) {
                const __bf16* p = Bsb + (waveN * 32 + tn * 16 + r16) * LDK + koff;
                bF[tn].q.lo = *(const int4*)p;
                bF[tn].q.hi = *(const int4*)(p + 16);
            }
#pragma unroll
            for (int tm = 0; tm < 4; ++tm) {
                const __bf16* p = Asb + (waveM * 64 + tm * 16 + r16) * LDK + koff;
                aF[tm].q.lo = *(const int4*)p;
                aF[tm].q.hi = *(const int4*)(p + 16);
            }
#pragma unroll
            for (int tm = 0; tm < 4; ++tm)
#pragma unroll
                for (int tn = 0; tn < 2; ++tn)
                    acc[tm][tn] = __builtin_amdgcn_wmma_f32_16x16x32_bf16(
                        false, aF[tm].v, false, bF[tn].v,
                        (short)0, acc[tm][tn], false, false);
        }
        __syncthreads();   // everyone done reading buf before it is refilled
    }

    // Epilogue. D layout: VGPR r -> M=r (lanes 0-15) / M=r+8 (lanes 16-31), N=lane&15.
#pragma unroll
    for (int tm = 0; tm < 4; ++tm) {
        const int srow0 = blockM0 + waveM * 64 + tm * 16 + half * 8;
#pragma unroll
        for (int tn = 0; tn < 2; ++tn) {
            const int ncol = blockN0 + waveN * 32 + tn * 16 + r16;
            const float sc = scale[ncol];
            if (EPI == 0) {
                __bf16* Y = (__bf16*)outp;
#pragma unroll
                for (int r = 0; r < 8; ++r)
                    Y[(size_t)(srow0 + r) * N_DIM + ncol] = (__bf16)(acc[tm][tn][r] * sc);
            } else {
                float* O = (float*)outp;
                const float bs = bias[ncol];
#pragma unroll
                for (int r = 0; r < 8; ++r)
                    O[(size_t)(srow0 + r) * N_DIM + ncol] = acc[tm][tn][r] * sc + bs;
            }
        }
    }
}

extern "C" void kernel_launch(void* const* d_in, const int* in_sizes, int n_in,
                              void* d_out, int out_size, void* d_ws, size_t ws_size,
                              hipStream_t stream) {
    const float* x    = (const float*)d_in[0];
    const float* s0   = (const float*)d_in[1];
    const float* B    = (const float*)d_in[2];
    const float* s2   = (const float*)d_in[3];
    const float* A    = (const float*)d_in[4];
    const float* s4   = (const float*)d_in[5];
    const float* bias = (const float*)d_in[6];

    char* ws = (char*)d_ws;
    __bf16* xs = (__bf16*)(ws);                                  // 64 MiB: bf16(x*s0)
    __bf16* Bb = (__bf16*)(ws + (size_t)64  * 1024 * 1024);      // 32 MiB: bf16(B)
    __bf16* Ab = (__bf16*)(ws + (size_t)96  * 1024 * 1024);      // 32 MiB: bf16(A)
    __bf16* Y  = (__bf16*)(ws + (size_t)128 * 1024 * 1024);      // 64 MiB: mid activation

    scale_cvt_x<<<(TOKENS * K_DIM) / (4 * 256), 256, 0, stream>>>(x, s0, xs);
    cvt_w<<<(N_DIM * K_DIM) / (4 * 256), 256, 0, stream>>>(B, Bb);
    cvt_w<<<(N_DIM * K_DIM) / (4 * 256), 256, 0, stream>>>(A, Ab);

    dim3 grid(N_DIM / 128, TOKENS / 128);
    gemm_bin_bf16<0><<<grid, 256, 0, stream>>>(xs, Bb, (void*)Y, s2, nullptr);
    gemm_bin_bf16<1><<<grid, 256, 0, stream>>>(Y, Ab, d_out, s4, bias);
}
